// LoRALinear_39324720562826
// MI455X (gfx1250) — matrix operand order repaired and checked by reading, not compiled, not run on hardware
//
#include <hip/hip_runtime.h>

// ---------------------------------------------------------------------------
// LoRA linear on MI455X (gfx1250):  y = x @ (W + B@A)^T + (b + lora_bias)
//
// Pipeline:
//  1) lora_prep : W_eff = W + B@A, split to bf16 hi/lo  -> ws   (runs once/call)
//  2) x_split   : x (f32) split to bf16 hi/lo           -> ws   (runs once/call)
//  3) lora_gemm : 8192x4096x4096 GEMM on the bf16 WMMA pipe with the
//     "3xBF16 ~= FP32" scheme (hi*hi + hi*lo + lo*hi, fp32 accum).
//     Hot loop is conversion-free: all operand tiles stream to LDS with
//     global_load_async_to_lds_b128 (double buffered), operands fetched with
//     ds_load_b128, 48 v_wmma per K-chunk. Compute-bound by design
//     (arith intensity ~820 FLOP/B >> 23.3 TB/s roofline knee).
// ---------------------------------------------------------------------------

typedef unsigned short u16;
typedef __attribute__((ext_vector_type(16))) __bf16 v16bf;
typedef __attribute__((ext_vector_type(8)))  __bf16 v8bf;
typedef __attribute__((ext_vector_type(8)))  float  v8f;
typedef __attribute__((ext_vector_type(4)))  u16    u16x4;
typedef __attribute__((ext_vector_type(4)))  float  f32x4;

// GCC-style vector matching the async builtin's parameter type
typedef int b128_t __attribute__((vector_size(16)));
typedef __attribute__((address_space(1))) b128_t* gptr_b128;
typedef __attribute__((address_space(3))) b128_t* lptr_b128;

#define KDIM 4096
#define NDIM 4096
#define BM 128
#define BN 128
#define BK 64
#define LDT 72                 // padded row stride in u16 (144B: 16B-aligned; 36-bank stride
                               // -> 16-row-strided b128 reads cover all 64 banks conflict-free)
#define TILE_U16 (BM * LDT)    // 9216
#define BUF_U16 (4 * TILE_U16) // xh, xl, wh, wl   (double buffered: 144 KB total LDS)

// ---- bf16 helpers (round-to-nearest-even) ----
__device__ __forceinline__ u16 f2bf_rne(float f) {
    union { float f; unsigned u; } v; v.f = f;
    unsigned u = v.u + 0x7fffu + ((v.u >> 16) & 1u);
    return (u16)(u >> 16);
}
__device__ __forceinline__ float bf2f(u16 h) {
    union { unsigned u; float f; } v; v.u = ((unsigned)h) << 16;
    return v.f;
}

// ---- async global->LDS copy (CDNA5) ----
__device__ __forceinline__ void async_ld_b128(const void* g, void* l) {
#if __has_builtin(__builtin_amdgcn_global_load_async_to_lds_b128)
    __builtin_amdgcn_global_load_async_to_lds_b128(
        (gptr_b128)(unsigned long long)(size_t)g,
        (lptr_b128)(unsigned)(size_t)l,
        0, 0);
#else
    unsigned           lds = (unsigned)(size_t)l;
    unsigned long long ga  = (unsigned long long)(size_t)g;
    asm volatile("global_load_async_to_lds_b128 %0, %1, off"
                 :: "v"(lds), "v"(ga) : "memory");
#endif
}
__device__ __forceinline__ void wait_async0() {
#if __has_builtin(__builtin_amdgcn_s_wait_asynccnt)
    __builtin_amdgcn_s_wait_asynccnt(0);
#else
    asm volatile("s_wait_asynccnt 0x0" ::: "memory");
#endif
}

// ---- LDS -> v16bf operand fetch ----
// A operand (16-bit 16x32 layout): halves at p and p+16 elements
__device__ __forceinline__ v16bf ld_split16(const u16* p) {
    v8bf a = *(const v8bf*)(p);
    v8bf b = *(const v8bf*)(p + 16);
    return __builtin_shufflevector(a, b, 0,1,2,3,4,5,6,7,8,9,10,11,12,13,14,15);
}
// 16 contiguous K values (unused: B is fetched with ld_split16 too)
__device__ __forceinline__ v16bf ld_contig16(const u16* p) {
    v8bf a = *(const v8bf*)(p);
    v8bf b = *(const v8bf*)(p + 8);
    return __builtin_shufflevector(a, b, 0,1,2,3,4,5,6,7,8,9,10,11,12,13,14,15);
}

// ---------------------------------------------------------------------------
// Prep 1: W_eff[n][k] = W[n][k] + sum_r B[n][r]*A[r][k]  (scaling = 1.0),
// stored as bf16 hi/lo pair, [N][K] row-major.
// ---------------------------------------------------------------------------
__launch_bounds__(256)
__global__ void lora_prep(const float* __restrict__ W,
                          const float* __restrict__ A,
                          const float* __restrict__ B,
                          u16* __restrict__ Whi, u16* __restrict__ Wlo) {
    const int n = blockIdx.x;
    __shared__ float Bn[16];
    if (threadIdx.x < 16) Bn[threadIdx.x] = B[n * 16 + threadIdx.x];
    __syncthreads();

    const int k0 = threadIdx.x * 16;
    float v[16];
#pragma unroll
    for (int j = 0; j < 16; j += 4) {
        f32x4 w = *(const f32x4*)(W + (size_t)n * KDIM + k0 + j);
        v[j] = w[0]; v[j+1] = w[1]; v[j+2] = w[2]; v[j+3] = w[3];
    }
#pragma unroll
    for (int r = 0; r < 16; ++r) {
        const float br = Bn[r];
#pragma unroll
        for (int j = 0; j < 16; j += 4) {
            f32x4 a = *(const f32x4*)(A + (size_t)r * KDIM + k0 + j);
            v[j]   += br * a[0]; v[j+1] += br * a[1];
            v[j+2] += br * a[2]; v[j+3] += br * a[3];
        }
    }
#pragma unroll
    for (int j = 0; j < 16; ++j) {
        u16 h = f2bf_rne(v[j]);
        Whi[(size_t)n * KDIM + k0 + j] = h;
        Wlo[(size_t)n * KDIM + k0 + j] = f2bf_rne(v[j] - bf2f(h));
    }
}

// ---------------------------------------------------------------------------
// Prep 2: split x (f32) into bf16 hi/lo arrays (one pass, bandwidth bound).
// ---------------------------------------------------------------------------
__launch_bounds__(256)
__global__ void x_split(const float* __restrict__ x,
                        u16* __restrict__ Xhi, u16* __restrict__ Xlo) {
    const size_t i = ((size_t)blockIdx.x * 256 + threadIdx.x) * 4;
    f32x4 v = *(const f32x4*)(x + i);
    u16x4 h, l;
#pragma unroll
    for (int j = 0; j < 4; ++j) {
        u16 hb = f2bf_rne(v[j]);
        h[j] = hb;
        l[j] = f2bf_rne(v[j] - bf2f(hb));
    }
    *(u16x4*)(Xhi + i) = h;
    *(u16x4*)(Xlo + i) = l;
}

// ---------------------------------------------------------------------------
// Main GEMM: y[M][N] = x[M][K] @ W_eff[N][K]^T + bias
// ---------------------------------------------------------------------------
__launch_bounds__(256)
__global__ void lora_gemm(const u16* __restrict__ Xhi,
                          const u16* __restrict__ Xlo,
                          const u16* __restrict__ Whi,
                          const u16* __restrict__ Wlo,
                          const float* __restrict__ bias,
                          const float* __restrict__ lbias,
                          float* __restrict__ y) {
    __shared__ __align__(16) u16 sm[2 * BUF_U16]; // 144 KB, double buffered

    const int tid  = threadIdx.x;
    const int lane = tid & 31;
    const int wid  = tid >> 5;
    const int wm   = wid & 3;   // 4 waves along M
    const int wn   = wid >> 2;  // 2 waves along N
    const int bm   = blockIdx.y * BM;
    const int bn   = blockIdx.x * BN;

    v8f acc[2][4];
#pragma unroll
    for (int i = 0; i < 2; ++i)
#pragma unroll
        for (int j = 0; j < 4; ++j)
            acc[i][j] = (v8f){0.f,0.f,0.f,0.f,0.f,0.f,0.f,0.f};

    // Stage all four 128x64-bf16 tiles purely with async copies
    // (1024 16B chunks per tile; 16 async instr per thread per chunk).
    auto stage = [&](int buf, int k0) {
        u16* sb = &sm[buf * BUF_U16];
#pragma unroll
        for (int c = 0; c < 4; ++c) {
            int idx = tid + c * 256;          // 0..1023
            int row = idx >> 3, seg = idx & 7;
            int    lo = row * LDT + seg * 8;
            size_t gx = (size_t)(bm + row) * KDIM + k0 + seg * 8;
            size_t gw = (size_t)(bn + row) * KDIM + k0 + seg * 8;
            async_ld_b128(Xhi + gx, sb + lo);
            async_ld_b128(Xlo + gx, sb + TILE_U16 + lo);
            async_ld_b128(Whi + gw, sb + 2 * TILE_U16 + lo);
            async_ld_b128(Wlo + gw, sb + 3 * TILE_U16 + lo);
        }
    };

    auto compute = [&](int buf) {
        const u16* xh = &sm[buf * BUF_U16];
        const u16* xl = xh + TILE_U16;
        const u16* wh = xh + 2 * TILE_U16;
        const u16* wl = xh + 3 * TILE_U16;

        const int am  = wm * 32 + (lane & 15);
        const int bnr = wn * 64 + (lane & 15);
        const int ak  = (lane & 16) ? 8 : 0;   // A: K {0..7,16..23} / {8..15,24..31}
        const int bk  = (lane & 16) ? 8 : 0;   // B: K {0..7,16..23} / {8..15,24..31}

#pragma unroll
        for (int ks = 0; ks < 2; ++ks) {       // two 32-wide K steps per BK=64 chunk
            const int ao = ks * 32 + ak;
            const int bo = ks * 32 + bk;
            v16bf ah[2], al[2];
#pragma unroll
            for (int ms = 0; ms < 2; ++ms) {
                ah[ms] = ld_split16(xh + (am + ms * 16) * LDT + ao);
                al[ms] = ld_split16(xl + (am + ms * 16) * LDT + ao);
            }
#pragma unroll
            for (int ns = 0; ns < 4; ++ns) {
                v16bf bh = ld_split16(wh + (bnr + ns * 16) * LDT + bo);
                v16bf bl = ld_split16(wl + (bnr + ns * 16) * LDT + bo);
#pragma unroll
                for (int ms = 0; ms < 2; ++ms) {
                    acc[ms][ns] = __builtin_amdgcn_wmma_f32_16x16x32_bf16(
                        false, ah[ms], false, bh, (short)0, acc[ms][ns], false, false);
                    acc[ms][ns] = __builtin_amdgcn_wmma_f32_16x16x32_bf16(
                        false, al[ms], false, bh, (short)0, acc[ms][ns], false, false);
                    acc[ms][ns] = __builtin_amdgcn_wmma_f32_16x16x32_bf16(
                        false, ah[ms], false, bl, (short)0, acc[ms][ns], false, false);
                }
            }
        }
    };

    const int KC = KDIM / BK; // 64 K-chunks
    stage(0, 0);
    wait_async0();
    __syncthreads();

    for (int kc = 0; kc < KC; ++kc) {
        const int  buf  = kc & 1;
        const bool more = (kc + 1) < KC;
        if (more) stage(buf ^ 1, (kc + 1) * BK);  // async, overlaps the 48 WMMAs below
        compute(buf);
        if (more) wait_async0();                  // next buffer landed during compute
        __syncthreads();
    }

    // Epilogue: C/D 32-bit 16x16 layout. lane&15 = N, VGPR i -> M = i (+8 for lanes>=16)
#pragma unroll
    for (int ns = 0; ns < 4; ++ns) {
        const int col = bn + wn * 64 + ns * 16 + (lane & 15);
        const float bb = bias[col] + lbias[col];
#pragma unroll
        for (int ms = 0; ms < 2; ++ms) {
            const int r0 = bm + wm * 32 + ms * 16 + ((lane & 16) ? 8 : 0);
#pragma unroll
            for (int i = 0; i < 8; ++i) {
                y[(size_t)(r0 + i) * NDIM + col] = acc[ms][ns][i] + bb;
            }
        }
    }
}

extern "C" void kernel_launch(void* const* d_in, const int* in_sizes, int n_in,
                              void* d_out, int out_size, void* d_ws, size_t ws_size,
                              hipStream_t stream) {
    const float* x     = (const float*)d_in[0];
    const float* W     = (const float*)d_in[1];
    const float* b     = (const float*)d_in[2];
    const float* lA    = (const float*)d_in[3];
    const float* lB    = (const float*)d_in[4];
    const float* lbias = (const float*)d_in[5];
    float*       y     = (float*)d_out;

    const int xtotal = in_sizes[0];      // M*K
    const int M      = xtotal / KDIM;    // 8192

    // workspace: [Whi | Wlo | Xhi | Xlo] bf16 (32+32+64+64 MB)
    u16* Whi = (u16*)d_ws;
    u16* Wlo = Whi + (size_t)NDIM * KDIM;
    u16* Xhi = Wlo + (size_t)NDIM * KDIM;
    u16* Xlo = Xhi + (size_t)M * KDIM;

    lora_prep<<<NDIM, 256, 0, stream>>>(W, lA, lB, Whi, Wlo);
    x_split<<<xtotal / 1024, 256, 0, stream>>>(x, Xhi, Xlo);
    lora_gemm<<<dim3(NDIM / BN, M / BM), 256, 0, stream>>>(Xhi, Xlo, Whi, Wlo, b, lbias, y);
}
